// RelativeMultiHeadAttn_79164837200745
// MI455X (gfx1250) — matrix operand order repaired and checked by reading, not compiled, not run on hardware
//
#include <hip/hip_runtime.h>
#include <math.h>

// ---------------------------------------------------------------------------
// Types for CDNA5 WMMA (wave32)
// ---------------------------------------------------------------------------
typedef __bf16 bf16x16 __attribute__((ext_vector_type(16)));
typedef __bf16 bf16x8  __attribute__((ext_vector_type(8)));
typedef float  f32x8   __attribute__((ext_vector_type(8)));

#define D_MODEL 1024
#define N_HEADS 16
#define D_K 64
#define BATCH 8
#define SEQ 512
#define RPAD 1024   // R = 1023 padded to 1024 (row 1023 zeroed)

__device__ inline f32x8 wmma_bf16(bf16x16 a, bf16x16 b, f32x8 c) {
    return __builtin_amdgcn_wmma_f32_16x16x32_bf16(false, a, false, b,
                                                   (short)0, c, false, false);
}

// A fragment: 16x32 bf16, lane m = l&15; lanes hold K-run kb..kb+15, kb = (l<16)?0:16
__device__ inline bf16x16 frag_A_bf16(const __bf16* base, int ld, int kc) {
    int l  = threadIdx.x & 31;
    int m  = l & 15;
    int kb = (l < 16) ? 0 : 16;
    const __bf16* p = base + (size_t)m * ld + kc + kb;
    bf16x8 lo = *(const bf16x8*)p;
    bf16x8 hi = *(const bf16x8*)(p + 8);
    return __builtin_shufflevector(lo, hi, 0,1,2,3,4,5,6,7,8,9,10,11,12,13,14,15);
}

// A fragment sourced from f32 LDS (probabilities), converted to bf16
__device__ inline bf16x16 frag_A_f32(const float* base, int ld, int kc) {
    int l  = threadIdx.x & 31;
    int m  = l & 15;
    int kb = (l < 16) ? 0 : 16;
    const float* p = base + m * ld + kc + kb;
    bf16x16 r;
#pragma unroll
    for (int j = 0; j < 8; ++j) r[j] = (__bf16)p[j];
#pragma unroll
    for (int j = 0; j < 8; ++j) r[8 + j] = (__bf16)p[8 + j];
    return r;
}

// B fragment: 32x16 bf16 from B^T (row-major [N][K]); lane n = l&15, K-run kb..kb+15, kb = (l<16)?0:16
__device__ inline bf16x16 frag_B_bf16(const __bf16* baseT, int ld, int kc) {
    int l  = threadIdx.x & 31;
    int n  = l & 15;
    int kb = (l < 16) ? 0 : 16;
    return *(const bf16x16*)(baseT + (size_t)n * ld + kc + kb);
}

// ---------------------------------------------------------------------------
// Prep kernels
// ---------------------------------------------------------------------------
__global__ void conv_bf16_kernel(const float* __restrict__ src, __bf16* __restrict__ dst, int n) {
    int i = blockIdx.x * blockDim.x + threadIdx.x;
    if (i < n) dst[i] = (__bf16)src[i];
}

// WT[f][c] = W[c][f], both 1024x1024
__global__ void transpose_bf16_kernel(const float* __restrict__ W, __bf16* __restrict__ WT) {
    int r = blockIdx.x;                  // source row
    for (int c = threadIdx.x; c < D_MODEL; c += blockDim.x)
        WT[(size_t)c * D_MODEL + r] = (__bf16)W[(size_t)r * D_MODEL + c];
}

// Sinusoidal positional embedding table, bf16, rows padded to RPAD (row >= 1023 zeroed)
__global__ void posemb_kernel(__bf16* __restrict__ P) {
    int r = blockIdx.x;                  // 0..1023
    float rp = (float)(SEQ - 1 - r);     // 511 .. -511
    bool valid = (r < (2 * SEQ - 1));
    for (int f = threadIdx.x; f < D_MODEL; f += blockDim.x) {
        float val = 0.f;
        if (valid) {
            int j = f >> 1;
            float freq = __expf(-(2.0f * (float)j / (float)D_MODEL) * 9.210340371976184f); // ln(10000)
            float ang = rp * freq;
            float sc = 0.1f * (float)D_MODEL; // 102.4
            val = ((f & 1) == 0 ? __sinf(ang) : __cosf(ang)) * sc;
        }
        P[(size_t)r * D_MODEL + f] = (__bf16)val;
    }
}

// ---------------------------------------------------------------------------
// Generic TN GEMM (A[M,K] row-major bf16, BT[N,K] row-major bf16) with
// mode-dependent writeout. 64x64 block tile, 4 waves, 16x64 per wave.
//  MODE 1: out[b][n][i][d]  (b=row/512, i=row%512, n=col%16, d=col/16)  bf16
//  MODE 2: out[b][n][d][i]                                              bf16
//  MODE 3: out[n][row][d]   (single batch, RPAD rows)                   bf16
//  MODE 4: outF[row][col] = acc + resid[row][col]                       f32
// ---------------------------------------------------------------------------
template <int MODE>
__global__ void gemm_tn_kernel(const __bf16* __restrict__ A, const __bf16* __restrict__ BT,
                               int M, int N, int K,
                               __bf16* __restrict__ outB, float* __restrict__ outF,
                               const float* __restrict__ resid) {
    int w    = threadIdx.x >> 5;
    int l    = threadIdx.x & 31;
    int row0 = blockIdx.y * 64 + w * 16;
    int col0 = blockIdx.x * 64;

    f32x8 acc0 = {0.f,0.f,0.f,0.f,0.f,0.f,0.f,0.f};
    f32x8 acc1 = acc0, acc2 = acc0, acc3 = acc0;

    const __bf16* Arow = A + (size_t)row0 * K;
    const __bf16* B0 = BT + (size_t)(col0 +  0) * K;
    const __bf16* B1 = BT + (size_t)(col0 + 16) * K;
    const __bf16* B2 = BT + (size_t)(col0 + 32) * K;
    const __bf16* B3 = BT + (size_t)(col0 + 48) * K;

    for (int kc = 0; kc < K; kc += 32) {
        bf16x16 a = frag_A_bf16(Arow, K, kc);
        acc0 = wmma_bf16(a, frag_B_bf16(B0, K, kc), acc0);
        acc1 = wmma_bf16(a, frag_B_bf16(B1, K, kc), acc1);
        acc2 = wmma_bf16(a, frag_B_bf16(B2, K, kc), acc2);
        acc3 = wmma_bf16(a, frag_B_bf16(B3, K, kc), acc3);
    }

    int nl = l & 15;
    int mh = (l >> 4) * 8;
    f32x8 accs[4] = {acc0, acc1, acc2, acc3};
#pragma unroll
    for (int c = 0; c < 4; ++c) {
        int col = col0 + c * 16 + nl;
#pragma unroll
        for (int v = 0; v < 8; ++v) {
            int row   = row0 + mh + v;
            float val = accs[c][v];
            if (MODE == 1) {
                int b = row >> 9, i = row & 511, n = col & 15, d = col >> 4;
                outB[(((size_t)(b * N_HEADS + n) * SEQ + i) * D_K) + d] = (__bf16)val;
            } else if (MODE == 2) {
                int b = row >> 9, i = row & 511, n = col & 15, d = col >> 4;
                outB[(((size_t)(b * N_HEADS + n) * D_K + d) * SEQ) + i] = (__bf16)val;
            } else if (MODE == 3) {
                int n = col & 15, d = col >> 4;
                outB[((size_t)(n * RPAD + row) * D_K) + d] = (__bf16)val;
            } else {
                size_t o = (size_t)row * N + col;
                outF[o] = val + resid[o];
            }
        }
    }
}

// ---------------------------------------------------------------------------
// Bias dots and head-weight sigmoid
// ---------------------------------------------------------------------------
__global__ void cbias_kernel(const __bf16* __restrict__ hk, const float* __restrict__ ub,
                             float* __restrict__ cb) {
    int idx = blockIdx.x * blockDim.x + threadIdx.x; // (b*16+n)*512 + j
    if (idx >= BATCH * N_HEADS * SEQ) return;
    const __bf16* row = hk + (size_t)idx * D_K;
    float s = 0.f;
#pragma unroll
    for (int d = 0; d < D_K; ++d) s += (float)row[d] * ub[d];
    cb[idx] = s;
}

__global__ void dbias_kernel(const __bf16* __restrict__ hr, const float* __restrict__ vb,
                             float* __restrict__ db) {
    int idx = blockIdx.x * blockDim.x + threadIdx.x; // n*1024 + r
    if (idx >= N_HEADS * RPAD) return;
    const __bf16* row = hr + (size_t)idx * D_K;
    float s = 0.f;
#pragma unroll
    for (int d = 0; d < D_K; ++d) s += (float)row[d] * vb[d];
    db[idx] = s;
}

__global__ void headw_kernel(const float* __restrict__ q, const float* __restrict__ head_w,
                             float* __restrict__ hw) {
    int row = blockIdx.x;                 // 0..4095 = b*512+i
    int tid = threadIdx.x;                // 128
    int h = tid & 15, part = tid >> 4;    // 8 parts x 128 elems
    float s = 0.f;
    const float* qr = q + (size_t)row * D_MODEL;
    for (int e = part * 128; e < part * 128 + 128; ++e)
        s += qr[e] * head_w[(size_t)e * N_HEADS + h];
    __shared__ float red[128];
    red[tid] = s;
    __syncthreads();
    if (tid < 16) {
        float t = 0.f;
#pragma unroll
        for (int p = 0; p < 8; ++p) t += red[p * 16 + tid];
        hw[(size_t)row * N_HEADS + tid] = 1.f / (1.f + __expf(-t));
    }
}

// ---------------------------------------------------------------------------
// Attention kernel: one workgroup per (b, head, 16-row query tile), 4 waves.
// ---------------------------------------------------------------------------
__global__ void __launch_bounds__(128)
attn_kernel(const __bf16* __restrict__ hq, const __bf16* __restrict__ hk,
            const __bf16* __restrict__ hvT, const __bf16* __restrict__ hr,
            const float* __restrict__ cbias, const float* __restrict__ dbias,
            const float* __restrict__ hw, const unsigned char* __restrict__ mask,
            __bf16* __restrict__ attn_out) {
    const int b  = blockIdx.z;
    const int n  = blockIdx.y;
    const int i0 = blockIdx.x * 16;
    const int w  = threadIdx.x >> 5;
    const int l  = threadIdx.x & 31;
    const int tid = threadIdx.x;

    __shared__ float logits[16][SEQ];
    __shared__ float relbuf[4][16][32];
    __shared__ float red[128];
    __shared__ float rowstat[16];

    const __bf16* Qb = hq + (((size_t)(b * N_HEADS + n) * SEQ) + i0) * D_K;
    const __bf16* Kb = hk + ((size_t)(b * N_HEADS + n) * SEQ) * D_K;
    const __bf16* Rb = hr + ((size_t)n * RPAD) * D_K;
    const float*  cbp = cbias + (size_t)(b * N_HEADS + n) * SEQ;
    const float*  dbp = dbias + (size_t)n * RPAD;

    bf16x16 aQ0 = frag_A_bf16(Qb, D_K, 0);
    bf16x16 aQ1 = frag_A_bf16(Qb, D_K, 32);

    const int nl = l & 15;
    const int mh = (l >> 4) * 8;
    const f32x8 z = {0.f,0.f,0.f,0.f,0.f,0.f,0.f,0.f};

    // ---- QK^T + relative term, 8 k-tiles per wave ----
    for (int kt = w; kt < SEQ / 16; kt += 4) {
        int k0 = kt * 16;
        f32x8 accC = z;
        accC = wmma_bf16(aQ0, frag_B_bf16(Kb + (size_t)k0 * D_K, D_K, 0),  accC);
        accC = wmma_bf16(aQ1, frag_B_bf16(Kb + (size_t)k0 * D_K, D_K, 32), accC);

        int c0 = (SEQ - 1) - i0 + k0 - 15;  // in [0, 992]; c0+31 <= 1023 (padded)
        f32x8 accR0 = z, accR1 = z;
        accR0 = wmma_bf16(aQ0, frag_B_bf16(Rb + (size_t)c0 * D_K, D_K, 0),          accR0);
        accR0 = wmma_bf16(aQ1, frag_B_bf16(Rb + (size_t)c0 * D_K, D_K, 32),         accR0);
        accR1 = wmma_bf16(aQ0, frag_B_bf16(Rb + (size_t)(c0 + 16) * D_K, D_K, 0),   accR1);
        accR1 = wmma_bf16(aQ1, frag_B_bf16(Rb + (size_t)(c0 + 16) * D_K, D_K, 32),  accR1);

#pragma unroll
        for (int v = 0; v < 8; ++v) {
            relbuf[w][mh + v][nl]      = accR0[v];
            relbuf[w][mh + v][16 + nl] = accR1[v];
        }
        // wave-internal LDS RAW: DS ops in-order; compiler inserts dscnt waits
#pragma unroll
        for (int v = 0; v < 8; ++v) {
            int m  = mh + v;
            int kg = k0 + nl;
            float rel = relbuf[w][m][15 + nl - m];
            float s = (accC[v] + rel + cbp[kg] + dbp[c0 + 15 + nl - m]) * 0.03125f;
            if (mask[((size_t)b * SEQ + (i0 + m)) * SEQ + kg]) s = -1e9f;
            logits[m][kg] = s;
        }
    }
    __syncthreads();

    // ---- softmax over k (512) for 16 rows; 8 slices of 64 per row ----
    {
        int r = tid & 15, sl = tid >> 4;
        float mx = -3.4e38f;
        for (int j = sl * 64; j < sl * 64 + 64; ++j) mx = fmaxf(mx, logits[r][j]);
        red[tid] = mx;
        __syncthreads();
        if (tid < 16) {
            float m2 = red[tid];
#pragma unroll
            for (int p = 1; p < 8; ++p) m2 = fmaxf(m2, red[p * 16 + tid]);
            rowstat[tid] = m2;
        }
        __syncthreads();
        float rmax = rowstat[r];
        float s = 0.f;
        for (int j = sl * 64; j < sl * 64 + 64; ++j) {
            float e = __expf(logits[r][j] - rmax);
            logits[r][j] = e;
            s += e;
        }
        red[tid] = s;
        __syncthreads();
        if (tid < 16) {
            float t = 0.f;
#pragma unroll
            for (int p = 0; p < 8; ++p) t += red[p * 16 + tid];
            rowstat[tid] = 1.f / t;
        }
        __syncthreads();
        float rinv = rowstat[r];
        for (int j = sl * 64; j < sl * 64 + 64; ++j) logits[r][j] *= rinv;
        __syncthreads();
    }

    // ---- O = P * V : wave w handles d-columns [16w, 16w+16) ----
    const __bf16* Vb = hvT + ((size_t)(b * N_HEADS + n) * D_K + w * 16) * SEQ;
    f32x8 accO = z;
    for (int kc = 0; kc < SEQ; kc += 32) {
        bf16x16 aP = frag_A_f32(&logits[0][0], SEQ, kc);
        bf16x16 bV = frag_B_bf16(Vb, SEQ, kc);
        accO = wmma_bf16(aP, bV, accO);
    }
#pragma unroll
    for (int v = 0; v < 8; ++v) {
        int m = mh + v;
        int i = i0 + m;
        int d = w * 16 + nl;
        float o = accO[v] * hw[((size_t)b * SEQ + i) * N_HEADS + n];
        attn_out[((size_t)b * SEQ + i) * D_MODEL + d * N_HEADS + n] = (__bf16)o;
    }
}

// ---------------------------------------------------------------------------
// In-place LayerNorm over 1024 features per row
// ---------------------------------------------------------------------------
__global__ void ln_kernel(float* __restrict__ out, const float* __restrict__ g,
                          const float* __restrict__ bta) {
    int row = blockIdx.x;
    int tid = threadIdx.x;
    float* rp = out + (size_t)row * D_MODEL;
    float s = 0.f, s2 = 0.f;
    for (int j = tid; j < D_MODEL; j += 256) {
        float x = rp[j];
        s += x; s2 += x * x;
    }
    __shared__ float r1[256], r2[256];
    __shared__ float mv[2];
    r1[tid] = s; r2[tid] = s2;
    __syncthreads();
    for (int st = 128; st > 0; st >>= 1) {
        if (tid < st) { r1[tid] += r1[tid + st]; r2[tid] += r2[tid + st]; }
        __syncthreads();
    }
    if (tid == 0) {
        float mean = r1[0] * (1.f / D_MODEL);
        float var  = r2[0] * (1.f / D_MODEL) - mean * mean;
        mv[0] = mean;
        mv[1] = rsqrtf(var + 1e-5f);
    }
    __syncthreads();
    float mean = mv[0], rstd = mv[1];
    for (int j = tid; j < D_MODEL; j += 256)
        rp[j] = (rp[j] - mean) * rstd * g[j] + bta[j];
}

// ---------------------------------------------------------------------------
// Host orchestration
// ---------------------------------------------------------------------------
extern "C" void kernel_launch(void* const* d_in, const int* in_sizes, int n_in,
                              void* d_out, int out_size, void* d_ws, size_t ws_size,
                              hipStream_t stream) {
    (void)in_sizes; (void)n_in; (void)out_size; (void)ws_size;

    const float* q      = (const float*)d_in[0];
    const float* k      = (const float*)d_in[1];
    const float* v      = (const float*)d_in[2];
    const unsigned char* mask = (const unsigned char*)d_in[3];
    const float* Wq     = (const float*)d_in[4];
    const float* Wk     = (const float*)d_in[5];
    const float* Wv     = (const float*)d_in[6];
    const float* Wr     = (const float*)d_in[7];
    const float* ub     = (const float*)d_in[8];
    const float* vb     = (const float*)d_in[9];
    const float* head_w = (const float*)d_in[10];
    const float* Wo     = (const float*)d_in[11];
    const float* ln_g   = (const float*)d_in[12];
    const float* ln_b   = (const float*)d_in[13];
    float* out = (float*)d_out;

    char* ws = (char*)d_ws;
    size_t off = 0;
    auto alloc = [&](size_t bytes) -> char* {
        char* p = ws + off;
        off += (bytes + 255) & ~(size_t)255;
        return p;
    };

    const size_t ROWS = (size_t)BATCH * SEQ;                 // 4096
    __bf16* qb  = (__bf16*)alloc(ROWS * D_MODEL * 2);
    __bf16* kb  = (__bf16*)alloc(ROWS * D_MODEL * 2);
    __bf16* vbb = (__bf16*)alloc(ROWS * D_MODEL * 2);
    __bf16* WqT = (__bf16*)alloc((size_t)D_MODEL * D_MODEL * 2);
    __bf16* WkT = (__bf16*)alloc((size_t)D_MODEL * D_MODEL * 2);
    __bf16* WvT = (__bf16*)alloc((size_t)D_MODEL * D_MODEL * 2);
    __bf16* WrT = (__bf16*)alloc((size_t)D_MODEL * D_MODEL * 2);
    __bf16* WoT = (__bf16*)alloc((size_t)D_MODEL * D_MODEL * 2);
    __bf16* Pb  = (__bf16*)alloc((size_t)RPAD * D_MODEL * 2);
    __bf16* hq  = (__bf16*)alloc(ROWS * D_MODEL * 2);        // [b][n][i][d]
    __bf16* hk  = (__bf16*)alloc(ROWS * D_MODEL * 2);        // [b][n][i][d]
    __bf16* hvT = (__bf16*)alloc(ROWS * D_MODEL * 2);        // [b][n][d][i]
    __bf16* hrp = (__bf16*)alloc((size_t)N_HEADS * RPAD * D_K * 2); // [n][r][d]
    float*  cb  = (float*)alloc((size_t)BATCH * N_HEADS * SEQ * 4);
    float*  db  = (float*)alloc((size_t)N_HEADS * RPAD * 4);
    float*  hw  = (float*)alloc(ROWS * N_HEADS * 4);
    __bf16* ao  = (__bf16*)alloc(ROWS * D_MODEL * 2);        // attention output [row][d*16+n]

    const int nElem = (int)(ROWS * D_MODEL);

    // 1) Convert activations to bf16
    conv_bf16_kernel<<<(nElem + 255) / 256, 256, 0, stream>>>(q, qb, nElem);
    conv_bf16_kernel<<<(nElem + 255) / 256, 256, 0, stream>>>(k, kb, nElem);
    conv_bf16_kernel<<<(nElem + 255) / 256, 256, 0, stream>>>(v, vbb, nElem);

    // 2) Transposed bf16 weights
    transpose_bf16_kernel<<<D_MODEL, 256, 0, stream>>>(Wq, WqT);
    transpose_bf16_kernel<<<D_MODEL, 256, 0, stream>>>(Wk, WkT);
    transpose_bf16_kernel<<<D_MODEL, 256, 0, stream>>>(Wv, WvT);
    transpose_bf16_kernel<<<D_MODEL, 256, 0, stream>>>(Wr, WrT);
    transpose_bf16_kernel<<<D_MODEL, 256, 0, stream>>>(Wo, WoT);

    // 3) Positional embedding table
    posemb_kernel<<<RPAD, 256, 0, stream>>>(Pb);

    // 4) Projections (WMMA)
    {
        dim3 grid(D_MODEL / 64, (unsigned)(ROWS / 64));
        gemm_tn_kernel<1><<<grid, 128, 0, stream>>>(qb, WqT, (int)ROWS, D_MODEL, D_MODEL, hq, nullptr, nullptr);
        gemm_tn_kernel<1><<<grid, 128, 0, stream>>>(kb, WkT, (int)ROWS, D_MODEL, D_MODEL, hk, nullptr, nullptr);
        gemm_tn_kernel<2><<<grid, 128, 0, stream>>>(vbb, WvT, (int)ROWS, D_MODEL, D_MODEL, hvT, nullptr, nullptr);
        dim3 gridR(D_MODEL / 64, RPAD / 64);
        gemm_tn_kernel<3><<<gridR, 128, 0, stream>>>(Pb, WrT, RPAD, D_MODEL, D_MODEL, hrp, nullptr, nullptr);
    }

    // 5) Bias dots + head weights
    cbias_kernel<<<(BATCH * N_HEADS * SEQ + 255) / 256, 256, 0, stream>>>(hk, ub, cb);
    dbias_kernel<<<(N_HEADS * RPAD + 255) / 256, 256, 0, stream>>>(hrp, vb, db);
    headw_kernel<<<(unsigned)ROWS, 128, 0, stream>>>(q, head_w, hw);

    // 6) Attention (WMMA QK^T + relative shift + softmax + WMMA PV)
    {
        dim3 grid(SEQ / 16, N_HEADS, BATCH);
        attn_kernel<<<grid, 128, 0, stream>>>(hq, hk, hvT, hrp, cb, db, hw, mask, ao);
    }

    // 7) Output projection + residual (WMMA), written to d_out
    {
        dim3 grid(D_MODEL / 64, (unsigned)(ROWS / 64));
        gemm_tn_kernel<4><<<grid, 128, 0, stream>>>(ao, WoT, (int)ROWS, D_MODEL, D_MODEL, nullptr, out, q);
    }

    // 8) In-place LayerNorm
    ln_kernel<<<(unsigned)ROWS, 256, 0, stream>>>(out, ln_g, ln_b);
}
